// _LinearAttention_81922206204651
// MI455X (gfx1250) — matrix operand order repaired and checked by reading, not compiled, not run on hardware
//
#include <hip/hip_runtime.h>

// ---- MI455X / gfx1250 linear attention (two-pass, f16 WMMA, f32 accum) ----

typedef __attribute__((ext_vector_type(16))) _Float16 v16h;
typedef __attribute__((ext_vector_type(8)))  _Float16 h8;
typedef __attribute__((ext_vector_type(8)))  float    v8f;
typedef __attribute__((ext_vector_type(4)))  float    f4;

#define S_LEN  4096
#define HEADS  8
#define DIM    64
#define ROWSTR (HEADS * DIM)   // 512 floats between consecutive s for fixed (n,h)
#define KT_STR 40              // LDS halfs per row: 32 data + 8 pad = 80B (16B multiple)
#define Q_STR  72              // LDS halfs per row: 64 data + 8 pad = 144B (16B multiple)

union V16 { v16h v; h8 h[2]; };

__device__ __forceinline__ float elu1(float x) {
    return x > 0.f ? x + 1.f : __expf(x);   // elu(x)+1
}

// ---------------------------------------------------------------------------
// Pass 1: KV[nh][d][v] += Kf^T @ V over a 256-row sequence chunk, Ksum[nh][d]
// grid = (32 nh, 16 chunks), block = 128 threads (4 waves)
// ---------------------------------------------------------------------------
__global__ __launch_bounds__(128)
void kv_kernel(const float* __restrict__ K, const float* __restrict__ V,
               float* __restrict__ kv_ws, float* __restrict__ ksum_ws) {
    __shared__ __align__(16) _Float16 KT[DIM][KT_STR];  // KT[d][s]  (A operand)
    __shared__ __align__(16) _Float16 VT[DIM][KT_STR];  // VT[v][s]  (B operand)

    const int tid  = threadIdx.x;
    const int lane = tid & 31;
    const int wave = tid >> 5;
    const int hi   = lane >> 4;      // 0/1: half-wave select
    const int lo   = lane & 15;
    const int nh   = blockIdx.x;
    const int n    = nh >> 3;
    const int head = nh & 7;
    const int s0   = blockIdx.y * 256;

    const float* kbase = K + ((size_t)n * S_LEN * HEADS + head) * DIM;
    const float* vbase = V + ((size_t)n * S_LEN * HEADS + head) * DIM;

    v8f acc[4] = {};                 // wave owns d-rows [16*wave, 16*wave+16), v = 0..63
    float ksum = 0.f;

    for (int sb = 0; sb < 8; ++sb) {
        const int sc = s0 + sb * 32;
        // stage 32 seq rows of Kf (elu+1, f16) and V (f16), transposed into LDS
        #pragma unroll
        for (int it = 0; it < 4; ++it) {
            const int i   = tid + it * 128;   // float4 index 0..511
            const int row = i >> 4;           // 0..31 (local s)
            const int d4  = (i & 15) << 2;    // 0..60
            const size_t g = (size_t)(sc + row) * ROWSTR + d4;
            const f4 kq = *(const f4*)(kbase + g);
            const f4 vq = *(const f4*)(vbase + g);
            #pragma unroll
            for (int j = 0; j < 4; ++j) {
                KT[d4 + j][row] = (_Float16)elu1(kq[j]);
                VT[d4 + j][row] = (_Float16)vq[j];
            }
        }
        __syncthreads();

        // Ksum partials (threads 0..63, one d column each)
        if (tid < DIM) {
            const _Float16* kr = &KT[tid][0];
            #pragma unroll
            for (int j = 0; j < 32; ++j) ksum += (float)kr[j];
        }

        // pack A (16x32 f16): lane<16 holds K=0..7 & 16..23 of row m; lane>=16 gets +8
        const _Float16* arow = &KT[16 * wave + lo][0];
        V16 a;
        a.h[0] = *(const h8*)(arow + 8 * hi);        // K = 8*hi + 0..7
        a.h[1] = *(const h8*)(arow + 8 * hi + 16);   // K = 8*hi + 16..23
        // 4 N-tiles of B (32x16 f16): lane<16 holds K=0..7 & 16..23; lane>=16 gets +8
        #pragma unroll
        for (int nt = 0; nt < 4; ++nt) {
            const _Float16* brow = &VT[nt * 16 + lo][0];
            V16 b;
            b.h[0] = *(const h8*)(brow + 8 * hi);
            b.h[1] = *(const h8*)(brow + 8 * hi + 16);
            acc[nt] = __builtin_amdgcn_wmma_f32_16x16x32_f16(
                false, a.v, false, b.v, (short)0, acc[nt], false, false);
        }
        __syncthreads();
    }

    // flush partial KV (f32 C layout: VGPR r -> M = r + 8*hi, N = lo)
    float* kvdst = kv_ws + (size_t)nh * DIM * DIM;
    #pragma unroll
    for (int nt = 0; nt < 4; ++nt) {
        #pragma unroll
        for (int r = 0; r < 8; ++r) {
            const int d = 16 * wave + 8 * hi + r;
            const int v = nt * 16 + lo;
            unsafeAtomicAdd(&kvdst[d * DIM + v], acc[nt][r]);
        }
    }
    if (tid < DIM) unsafeAtomicAdd(&ksum_ws[nh * DIM + tid], ksum);
}

// ---------------------------------------------------------------------------
// Pass 2: out[l,:] = (Qf[l,:] @ KV) * 1/(Qf[l,:]·Ksum + eps)
// grid = (32 nh, 64 l-tiles of 64 rows), block = 128 threads (4 waves)
// ---------------------------------------------------------------------------
__global__ __launch_bounds__(128)
void out_kernel(const float* __restrict__ Q, const float* __restrict__ kv_ws,
                const float* __restrict__ ksum_ws, float* __restrict__ out) {
    __shared__ __align__(16) _Float16 KVT[DIM][Q_STR];  // KVT[v][d] (B operand)
    __shared__ __align__(16) _Float16 QL[64][Q_STR];    // QL[l][d]  (A operand, natural)
    __shared__ float KS[DIM];

    const int tid  = threadIdx.x;
    const int lane = tid & 31;
    const int wave = tid >> 5;
    const int hi   = lane >> 4;
    const int lo   = lane & 15;
    const int nh   = blockIdx.x;
    const int n    = nh >> 3;
    const int head = nh & 7;
    const int l0   = blockIdx.y * 64;

    // stage KV (f32 -> f16, transposed) and Ksum
    const float* kvsrc = kv_ws + (size_t)nh * DIM * DIM;
    for (int i = tid; i < DIM * DIM; i += 128) {
        const int d = i >> 6, v = i & 63;
        KVT[v][d] = (_Float16)kvsrc[i];
    }
    if (tid < DIM) KS[tid] = ksum_ws[nh * DIM + tid];

    // stage 64 rows of Qf
    const float* qbase = Q + ((size_t)n * S_LEN * HEADS + head) * DIM;
    #pragma unroll
    for (int it = 0; it < 8; ++it) {
        const int i   = tid + it * 128;   // float4 index 0..1023
        const int row = i >> 4;           // 0..63
        const int d4  = (i & 15) << 2;
        const f4 qq = *(const f4*)(qbase + (size_t)(l0 + row) * ROWSTR + d4);
        #pragma unroll
        for (int j = 0; j < 4; ++j) QL[row][d4 + j] = (_Float16)elu1(qq[j]);
    }
    __syncthreads();

    // Z per row: lane (hi,lo) sums half the dot(Qf_row, Ksum), combine across half-waves
    const int rowl = 16 * wave + lo;
    const _Float16* qrow = &QL[rowl][0];
    float part = 0.f;
    #pragma unroll
    for (int j = 0; j < 32; ++j) part += (float)qrow[32 * hi + j] * KS[32 * hi + j];
    part += __shfl_xor(part, 16, 32);
    const float z = 1.f / (part + 1e-6f);

    // out tile: 2 K-steps (d = 0..31, 32..63) x 4 N-tiles
    v8f acc[4] = {};
    #pragma unroll
    for (int ks = 0; ks < 2; ++ks) {
        V16 a;
        a.h[0] = *(const h8*)(qrow + 32 * ks + 8 * hi);
        a.h[1] = *(const h8*)(qrow + 32 * ks + 8 * hi + 16);
        #pragma unroll
        for (int nt = 0; nt < 4; ++nt) {
            const _Float16* brow = &KVT[nt * 16 + lo][0];
            V16 b;
            b.h[0] = *(const h8*)(brow + 32 * ks + 8 * hi);
            b.h[1] = *(const h8*)(brow + 32 * ks + 8 * hi + 16);
            acc[nt] = __builtin_amdgcn_wmma_f32_16x16x32_f16(
                false, a.v, false, b.v, (short)0, acc[nt], false, false);
        }
    }

    // scale by Z and store (C layout: VGPR r -> row r + 8*hi, col lo)
    float* obase = out + ((size_t)n * S_LEN * HEADS + head) * DIM;
    #pragma unroll
    for (int r = 0; r < 8; ++r) {
        const float zr = __shfl(z, 8 * hi + r, 32);
        const size_t lrow = (size_t)(l0 + 16 * wave + 8 * hi + r) * ROWSTR;
        #pragma unroll
        for (int nt = 0; nt < 4; ++nt)
            obase[lrow + nt * 16 + lo] = acc[nt][r] * zr;
    }
}

// ---------------------------------------------------------------------------
extern "C" void kernel_launch(void* const* d_in, const int* in_sizes, int n_in,
                              void* d_out, int out_size, void* d_ws, size_t ws_size,
                              hipStream_t stream) {
    const float* Q = (const float*)d_in[0];
    const float* K = (const float*)d_in[1];
    const float* V = (const float*)d_in[2];
    float* out = (float*)d_out;

    float* kv_ws   = (float*)d_ws;                 // 32 * 64 * 64 f32 = 512 KB
    float* ksum_ws = kv_ws + 32 * DIM * DIM;       // 32 * 64 f32     =   8 KB
    const size_t ws_bytes = (size_t)(32 * DIM * DIM + 32 * DIM) * sizeof(float);

    hipMemsetAsync(d_ws, 0, ws_bytes, stream);     // atomics accumulate -> zero each call
    hipLaunchKernelGGL(kv_kernel, dim3(32, 16), dim3(128), 0, stream, K, V, kv_ws, ksum_ws);
    hipLaunchKernelGGL(out_kernel, dim3(32, 64), dim3(128), 0, stream, Q, kv_ws, ksum_ws, out);
}
